// Next_Node_Probability_Calculator_for_group_49340584297074
// MI455X (gfx1250) — matrix operand order repaired and checked by reading, not compiled, not run on hardware
//
#include <hip/hip_runtime.h>
#include <hip/hip_bf16.h>
#include <cstdint>
#include <cstddef>

typedef _Float16 half_t;
typedef __attribute__((ext_vector_type(16))) _Float16 v16h;
typedef __attribute__((ext_vector_type(8)))  _Float16 v8h;
typedef __attribute__((ext_vector_type(8)))  float    v8f;

#define DEV static __device__ __forceinline__

// Problem constants (match reference)
#define CB 32
#define CN 512
#define CG 512
#define CD 512
#define CH 16
#define CKD 32

// ---------------------------------------------------------------------------
// WMMA fragment helpers (CDNA5 wave32 16x16x32 f16 layouts, CDNA5 ISA)
//
// A-matrix 16x32 (MxK), lane l: m = l&15, hi = l>>4:
//   halfs j=0..7  <-> K = j + 8*hi          (contiguous 8)
//   halfs j=8..15 <-> K = 16 + (j-8) + 8*hi (contiguous 8)
// B-matrix 32x16 (KxN), lane l: n = l&15, hi = l>>4:
//   halfs j=0..7  <-> K = j + 8*hi; j=8..15 <-> K = 16 + (j-8) + 8*hi
// C/D 16x16 f32: vgpr r, lane l: row m = r + 8*hi, col n = l&15
// ---------------------------------------------------------------------------

DEV v16h load_a_frag(const half_t* __restrict__ row, int hi) {
  v8h lo = *(const v8h*)(row + 8 * hi);
  v8h hh = *(const v8h*)(row + 16 + 8 * hi);
  v16h a;
#pragma unroll
  for (int i = 0; i < 8; ++i) { a[i] = lo[i]; a[i + 8] = hh[i]; }
  return a;
}

DEV v16h load_a_frag_lds(const half_t* row, int hi) {
  v8h lo = *(const v8h*)(row + 8 * hi);
  v8h hh = *(const v8h*)(row + 16 + 8 * hi);
  v16h a;
#pragma unroll
  for (int i = 0; i < 8; ++i) { a[i] = lo[i]; a[i + 8] = hh[i]; }
  return a;
}

DEV v16h load_b_frag(const half_t* __restrict__ col, int hi) {
  v8h x0 = *(const v8h*)(col + 8 * hi);
  v8h x1 = *(const v8h*)(col + 16 + 8 * hi);
  v16h b;
#pragma unroll
  for (int i = 0; i < 8; ++i) { b[i] = x0[i]; b[i + 8] = x1[i]; }
  return b;
}

DEV v16h load_b_frag_lds(const half_t* col, int hi) {
  v8h x0 = *(const v8h*)(col + 8 * hi);
  v8h x1 = *(const v8h*)(col + 16 + 8 * hi);
  v16h b;
#pragma unroll
  for (int i = 0; i < 8; ++i) { b[i] = x0[i]; b[i + 8] = x1[i]; }
  return b;
}

DEV v8f wmma16(v16h a, v16h b, v8f c) {
  return __builtin_amdgcn_wmma_f32_16x16x32_f16(false, a, false, b, (short)0, c,
                                                false, false);
}

// ---------------------------------------------------------------------------
// Elementwise prep kernels
// ---------------------------------------------------------------------------
__global__ void cvt_f16_kernel(const float* __restrict__ src,
                               half_t* __restrict__ dst, int n) {
  int i = blockIdx.x * 256 + threadIdx.x;
  if (i < n) dst[i] = (half_t)src[i];
}

__global__ void addcvt_f16_kernel(const float* __restrict__ a,
                                  const float* __restrict__ b,
                                  half_t* __restrict__ dst, int n) {
  int i = blockIdx.x * 256 + threadIdx.x;
  if (i < n) dst[i] = (half_t)(a[i] + b[i]);
}

// gmean[b][d] = mean_n enc[b][n][d]
__global__ void mean_kernel(const float* __restrict__ enc,
                            float* __restrict__ gmean) {
  int i = blockIdx.x * 256 + threadIdx.x;  // i = b*D + d
  int b = i >> 9, d = i & 511;
  const float* p = enc + (size_t)b * CN * CD + d;
  float s = 0.f;
#pragma unroll 4
  for (int n = 0; n < CN; ++n) s += p[(size_t)n * CD];
  gmean[i] = s * (1.0f / (float)CN);
}

// gq[b][d] = sum_k gmean[b][k] * Wq_graph[d][k]
__global__ void graphq_kernel(const float* __restrict__ gmean,
                              const float* __restrict__ Wqg,
                              float* __restrict__ gq) {
  int i = blockIdx.x * 256 + threadIdx.x;  // i = b*D + d
  int b = i >> 9, d = i & 511;
  const float* g = gmean + (size_t)b * CD;
  const float* w = Wqg + (size_t)d * CD;
  float s = 0.f;
#pragma unroll 4
  for (int k = 0; k < CD; ++k) s += g[k] * w[k];
  gq[i] = s;
}

// ---------------------------------------------------------------------------
// Unified WMMA GEMM: C[M,N] = X[M,K] @ W[N,K]^T  (all row-major, f16 in, f32 acc)
// Block: 256 threads = 8 waves; each wave owns a 32x64 tile (256 rows x 64 cols
// per block, all waves share the same 64-col W slice for L0/L2 reuse).
// Per 32-wide k-step: 4 A-loads + 8 B-loads feed 8 WMMAs (B frags reused 2x).
// mode 0: store f16
// mode 1: v = (v + extra[(row/rowsPerBatch)*N + col]) * scale; store f16 (Q path)
// mode 2: v = v + extra[col]; store f16 (bias)
// mode 3: v = v * scale; store f32 (batched stage-2 scores)
// ---------------------------------------------------------------------------
__global__ __launch_bounds__(256) void gemm_xwt_kernel(
    const half_t* __restrict__ X, const half_t* __restrict__ W,
    half_t* __restrict__ outH, float* __restrict__ outF,
    const float* __restrict__ extra, int Ncols, int Kdim, int mode, float scale,
    unsigned rowsPerBatch, size_t bsX, size_t bsW, size_t bsC) {
  const int lane = threadIdx.x & 31;
  const int wave = threadIdx.x >> 5;
  const int m16 = lane & 15, hi = lane >> 4;
  const int z = blockIdx.z;

  const half_t* Xb = X + (size_t)z * bsX;
  const half_t* Wb = W + (size_t)z * bsW;
  const int row0 = blockIdx.y * 256 + wave * 32;
  const int col0 = blockIdx.x * 64;

  const half_t* xrow0 = Xb + (size_t)(row0 + m16) * Kdim;
  const half_t* xrow1 = Xb + (size_t)(row0 + 16 + m16) * Kdim;
  const half_t* w0 = Wb + (size_t)(col0 + 0 + m16) * Kdim;
  const half_t* w1 = Wb + (size_t)(col0 + 16 + m16) * Kdim;
  const half_t* w2 = Wb + (size_t)(col0 + 32 + m16) * Kdim;
  const half_t* w3 = Wb + (size_t)(col0 + 48 + m16) * Kdim;

  v8f c0[4] = {v8f{}, v8f{}, v8f{}, v8f{}};   // rows row0 .. row0+15
  v8f c1[4] = {v8f{}, v8f{}, v8f{}, v8f{}};   // rows row0+16 .. row0+31

#pragma unroll 2
  for (int kt = 0; kt < Kdim; kt += 32) {
    __builtin_prefetch(xrow0 + kt + 128, 0, 1);  // global_prefetch_b8
    __builtin_prefetch(xrow1 + kt + 128, 0, 1);
    __builtin_prefetch(w0 + kt + 128, 0, 1);
    v16h a0 = load_a_frag(xrow0 + kt, hi);
    v16h a1 = load_a_frag(xrow1 + kt, hi);
    {
      v16h bf = load_b_frag(w0 + kt, hi);
      c0[0] = wmma16(a0, bf, c0[0]);
      c1[0] = wmma16(a1, bf, c1[0]);
    }
    {
      v16h bf = load_b_frag(w1 + kt, hi);
      c0[1] = wmma16(a0, bf, c0[1]);
      c1[1] = wmma16(a1, bf, c1[1]);
    }
    {
      v16h bf = load_b_frag(w2 + kt, hi);
      c0[2] = wmma16(a0, bf, c0[2]);
      c1[2] = wmma16(a1, bf, c1[2]);
    }
    {
      v16h bf = load_b_frag(w3 + kt, hi);
      c0[3] = wmma16(a0, bf, c0[3]);
      c1[3] = wmma16(a1, bf, c1[3]);
    }
  }

#pragma unroll
  for (int p = 0; p < 2; ++p) {
#pragma unroll
    for (int f = 0; f < 4; ++f) {
#pragma unroll
      for (int r = 0; r < 8; ++r) {
        unsigned grow = (unsigned)(row0 + p * 16 + r + 8 * hi);
        int gcol = col0 + f * 16 + m16;
        float v = (p == 0) ? c0[f][r] : c1[f][r];
        if (mode == 1) {
          v = (v + extra[(grow / rowsPerBatch) * (unsigned)Ncols + gcol]) *
              scale;
        } else if (mode == 2) {
          v = v + extra[gcol];
        } else if (mode == 3) {
          v = v * scale;
        }
        size_t idx = (size_t)grow * (size_t)Ncols + (size_t)gcol;
        if (mode == 3)
          outF[(size_t)z * bsC + idx] = v;
        else
          outH[idx] = (half_t)v;
      }
    }
  }
}

// ---------------------------------------------------------------------------
// Attention: per wave, one (b, h, 16-row G tile). Flash-style online softmax.
// Q was pre-scaled by 1/sqrt(KD).  32 nodes per step: 2 score WMMAs + 2 PV WMMAs.
// Per-wave LDS slice: vbufT[32 cols][32 nodes] f16 + pbuf[16 rows][32 nodes] f16.
// ---------------------------------------------------------------------------
__global__ __launch_bounds__(256) void attn_kernel(
    const half_t* __restrict__ Q, const half_t* __restrict__ Kh,
    const half_t* __restrict__ Vh, const float* __restrict__ mask,
    half_t* __restrict__ outAttn) {
  const int lane = threadIdx.x & 31;
  const int wave = threadIdx.x >> 5;
  const int w = blockIdx.x * 8 + wave;  // 0..16383
  const int gt = w & 31;                // G tile (16 rows each)
  const int h = (w >> 5) & 15;          // head
  const int b = w >> 9;                 // batch
  const int m16 = lane & 15, hi = lane >> 4;

  __shared__ half_t lds[8 * (32 * 32 + 16 * 32)];
  half_t* vbufT = lds + wave * (32 * 32 + 16 * 32);  // [col][node]
  half_t* pbuf = vbufT + 32 * 32;                    // [row][node]

  const half_t* qrow = Q + ((size_t)(b * CG + gt * 16 + m16) * CD + h * CKD);
  v16h aq = load_a_frag(qrow, hi);

  v8f acc0 = v8f{}, acc1 = v8f{};
  float Mrow[8], Lrow[8];
#pragma unroll
  for (int r = 0; r < 8; ++r) { Mrow[r] = -3.0e38f; Lrow[r] = 0.f; }

  const float* mbase = mask + (size_t)(b * CG + gt * 16) * CN;

  for (int nb = 0; nb < CN; nb += 32) {
    // Stage V[nb+lane][h*32 .. +32] transposed into vbufT[col][node]
    const half_t* vrow = Vh + ((size_t)(b * CN + nb + lane) * CD + h * CKD);
    v8h v0 = *(const v8h*)(vrow);
    v8h v1 = *(const v8h*)(vrow + 8);
    v8h v2 = *(const v8h*)(vrow + 16);
    v8h v3 = *(const v8h*)(vrow + 24);
#pragma unroll
    for (int cI = 0; cI < 8; ++cI) {
      vbufT[(cI + 0) * 32 + lane] = v0[cI];
      vbufT[(cI + 8) * 32 + lane] = v1[cI];
      vbufT[(cI + 16) * 32 + lane] = v2[cI];
      vbufT[(cI + 24) * 32 + lane] = v3[cI];
    }

    // scores for two 16-node tiles
    const half_t* krow0 = Kh + ((size_t)(b * CN + nb + m16) * CD + h * CKD);
    const half_t* krow1 = Kh + ((size_t)(b * CN + nb + 16 + m16) * CD + h * CKD);
    v8f s0 = v8f{}, s1 = v8f{};
    s0 = wmma16(aq, load_b_frag(krow0, hi), s0);
    s1 = wmma16(aq, load_b_frag(krow1, hi), s1);

    // mask add: element (row = r+8*hi, n = nb + m16 [+16])
    const float* mrow = mbase + nb + m16;
#pragma unroll
    for (int r = 0; r < 8; ++r) {
      int row = r + 8 * hi;
      s0[r] += mrow[(size_t)row * CN];
      s1[r] += mrow[(size_t)row * CN + 16];
    }

    // online softmax (row stats replicated across each 16-lane half)
#pragma unroll
    for (int r = 0; r < 8; ++r) {
      float t = fmaxf(s0[r], s1[r]);
      t = fmaxf(t, __shfl_xor(t, 1, 32));
      t = fmaxf(t, __shfl_xor(t, 2, 32));
      t = fmaxf(t, __shfl_xor(t, 4, 32));
      t = fmaxf(t, __shfl_xor(t, 8, 32));
      float newM = fmaxf(Mrow[r], t);
      float sc = __expf(Mrow[r] - newM);
      Mrow[r] = newM;
      float p0 = __expf(s0[r] - newM);
      float p1 = __expf(s1[r] - newM);
      s0[r] = p0;
      s1[r] = p1;
      float rs = p0 + p1;
      rs += __shfl_xor(rs, 1, 32);
      rs += __shfl_xor(rs, 2, 32);
      rs += __shfl_xor(rs, 4, 32);
      rs += __shfl_xor(rs, 8, 32);
      Lrow[r] = Lrow[r] * sc + rs;
      acc0[r] *= sc;
      acc1[r] *= sc;
    }

    // P tile -> LDS [16 rows][32 local nodes] f16
#pragma unroll
    for (int r = 0; r < 8; ++r) {
      int row = r + 8 * hi;
      pbuf[row * 32 + m16] = (half_t)s0[r];
      pbuf[row * 32 + 16 + m16] = (half_t)s1[r];
    }

    // P (A frag, K = 32 local nodes) x V (B frags from transposed LDS)
    v16h ap = load_a_frag_lds(pbuf + m16 * 32, hi);
    v16h bv0 = load_b_frag_lds(vbufT + m16 * 32, hi);
    v16h bv1 = load_b_frag_lds(vbufT + (16 + m16) * 32, hi);
    acc0 = wmma16(ap, bv0, acc0);
    acc1 = wmma16(ap, bv1, acc1);
  }

  // normalize + store f16 attention output [B*G, D]
  half_t* orow = outAttn + ((size_t)(b * CG + gt * 16) * CD + h * CKD);
#pragma unroll
  for (int r = 0; r < 8; ++r) {
    int row = r + 8 * hi;
    float inv = 1.0f / Lrow[r];
    orow[(size_t)row * CD + m16] = (half_t)(acc0[r] * inv);
    orow[(size_t)row * CD + 16 + m16] = (half_t)(acc1[r] * inv);
  }
}

// ---------------------------------------------------------------------------
// In-place: s2 -> softmax(10*tanh(s2) + mask) along last dim (512).
// One wave per row, 16 elements per lane.
// ---------------------------------------------------------------------------
__global__ __launch_bounds__(256) void clip_mask_softmax_kernel(
    float* __restrict__ s2, const float* __restrict__ mask) {
  const int lane = threadIdx.x & 31;
  const int wave = threadIdx.x >> 5;
  const size_t row = (size_t)blockIdx.x * 8 + wave;  // 0..16383
  float* p = s2 + row * CN;
  const float* mp = mask + row * CN;

  float v[16];
  float mx = -3.0e38f;
#pragma unroll
  for (int i = 0; i < 16; ++i) {
    int idx = lane + i * 32;
    float t = 10.0f * tanhf(p[idx]) + mp[idx];
    v[i] = t;
    mx = fmaxf(mx, t);
  }
  mx = fmaxf(mx, __shfl_xor(mx, 1, 32));
  mx = fmaxf(mx, __shfl_xor(mx, 2, 32));
  mx = fmaxf(mx, __shfl_xor(mx, 4, 32));
  mx = fmaxf(mx, __shfl_xor(mx, 8, 32));
  mx = fmaxf(mx, __shfl_xor(mx, 16, 32));

  float s = 0.f;
#pragma unroll
  for (int i = 0; i < 16; ++i) {
    v[i] = __expf(v[i] - mx);
    s += v[i];
  }
  s += __shfl_xor(s, 1, 32);
  s += __shfl_xor(s, 2, 32);
  s += __shfl_xor(s, 4, 32);
  s += __shfl_xor(s, 8, 32);
  s += __shfl_xor(s, 16, 32);
  float inv = 1.0f / s;
#pragma unroll
  for (int i = 0; i < 16; ++i) p[lane + i * 32] = v[i] * inv;
}

// ---------------------------------------------------------------------------
// Host launcher
// ---------------------------------------------------------------------------
extern "C" void kernel_launch(void* const* d_in, const int* in_sizes, int n_in,
                              void* d_out, int out_size, void* d_ws,
                              size_t ws_size, hipStream_t stream) {
  (void)in_sizes; (void)n_in; (void)out_size; (void)ws_size;

  const float* enc_nodes = (const float*)d_in[0];
  const float* enc_last = (const float*)d_in[1];
  const float* mask = (const float*)d_in[2];
  const float* Wq_graph = (const float*)d_in[3];
  const float* Wq_first = (const float*)d_in[4];
  const float* Wq_last = (const float*)d_in[5];
  const float* Wk = (const float*)d_in[6];
  const float* Wv = (const float*)d_in[7];
  const float* Wcomb = (const float*)d_in[8];
  const float* bcomb = (const float*)d_in[9];
  float* out = (float*)d_out;

  uint8_t* ws = (uint8_t*)d_ws;
  size_t off = 0;
  auto alloc = [&](size_t bytes) -> void* {
    void* p = ws + off;
    off += (bytes + 255) & ~(size_t)255;
    return p;
  };

  const size_t nBND = (size_t)CB * CN * CD;  // 8388608
  const size_t nDD = (size_t)CD * CD;        // 262144

  half_t* Xn_h = (half_t*)alloc(nBND * 2);   // encoded_nodes f16
  half_t* Xl_h = (half_t*)alloc(nBND * 2);   // encoded_last f16
  half_t* Wk_h = (half_t*)alloc(nDD * 2);
  half_t* Wv_h = (half_t*)alloc(nDD * 2);
  half_t* Wqs_h = (half_t*)alloc(nDD * 2);   // Wq_first + Wq_last
  half_t* Wc_h = (half_t*)alloc(nDD * 2);
  float* gmean = (float*)alloc((size_t)CB * CD * 4);
  float* gq = (float*)alloc((size_t)CB * CD * 4);
  half_t* Qh = (half_t*)alloc(nBND * 2);
  half_t* Kh = (half_t*)alloc(nBND * 2);
  half_t* Vh = (half_t*)alloc(nBND * 2);
  half_t* Ah = (half_t*)alloc(nBND * 2);     // attention output
  half_t* Mh = (half_t*)alloc(nBND * 2);     // mh (after Wcomb)

  // --- prep: f16 conversions ---
  int nb1 = (int)((nBND + 255) / 256);
  cvt_f16_kernel<<<nb1, 256, 0, stream>>>(enc_nodes, Xn_h, (int)nBND);
  cvt_f16_kernel<<<nb1, 256, 0, stream>>>(enc_last, Xl_h, (int)nBND);
  int nbw = (int)((nDD + 255) / 256);
  cvt_f16_kernel<<<nbw, 256, 0, stream>>>(Wk, Wk_h, (int)nDD);
  cvt_f16_kernel<<<nbw, 256, 0, stream>>>(Wv, Wv_h, (int)nDD);
  cvt_f16_kernel<<<nbw, 256, 0, stream>>>(Wcomb, Wc_h, (int)nDD);
  addcvt_f16_kernel<<<nbw, 256, 0, stream>>>(Wq_first, Wq_last, Wqs_h, (int)nDD);

  // --- graph query term ---
  mean_kernel<<<(CB * CD) / 256, 256, 0, stream>>>(enc_nodes, gmean);
  graphq_kernel<<<(CB * CD) / 256, 256, 0, stream>>>(gmean, Wq_graph, gq);

  const float invSqrtKD = 0.17677669529663689f;  // 1/sqrt(32)
  const float invSqrtD = 0.044194173824159216f;  // 1/sqrt(512)

  // --- projections: K, V, Q (Q folds graph term + 1/sqrt(KD)) ---
  dim3 gProj(CD / 64, (CB * CN) / 256, 1);
  gemm_xwt_kernel<<<gProj, 256, 0, stream>>>(Xn_h, Wk_h, Kh, nullptr, nullptr,
                                             CD, CD, 0, 1.0f, 1u, 0, 0, 0);
  gemm_xwt_kernel<<<gProj, 256, 0, stream>>>(Xn_h, Wv_h, Vh, nullptr, nullptr,
                                             CD, CD, 0, 1.0f, 1u, 0, 0, 0);
  gemm_xwt_kernel<<<gProj, 256, 0, stream>>>(Xl_h, Wqs_h, Qh, nullptr, gq, CD,
                                             CD, 1, invSqrtKD, (unsigned)CG, 0,
                                             0, 0);

  // --- stage 1 attention ---
  attn_kernel<<<(CB * CH * (CG / 16)) / 8, 256, 0, stream>>>(Qh, Kh, Vh, mask,
                                                             Ah);

  // --- mh = attn @ Wcomb^T + bcomb ---
  gemm_xwt_kernel<<<gProj, 256, 0, stream>>>(Ah, Wc_h, Mh, nullptr, bcomb, CD,
                                             CD, 2, 1.0f, 1u, 0, 0, 0);

  // --- stage 2 scores: per-batch  s2[b] = (mh[b] @ enc_nodes[b]^T)/sqrt(D)
  //     written directly into d_out as f32 ---
  dim3 gS2(CN / 64, CG / 256, CB);
  gemm_xwt_kernel<<<gS2, 256, 0, stream>>>(
      Mh, Xn_h, nullptr, out, nullptr, CN, CD, 3, invSqrtD, 1u,
      (size_t)CG * CD, (size_t)CN * CD, (size_t)CG * CN);

  // --- tanh clip + mask + softmax, in place on d_out ---
  clip_mask_softmax_kernel<<<(CB * CG) / 8, 256, 0, stream>>>(out, mask);
}
